// SimMIM_86131274154407
// MI455X (gfx1250) — matrix operand order repaired and checked by reading, not compiled, not run on hardware
//
#include <hip/hip_runtime.h>
#include <cstdint>
#include <cstddef>

// ---------------- problem constants ----------------
#define IMGSZ 224
#define PATCH 16
#define PGRID 14          // 224/16
#define NTOK  196         // patches per image
#define DIM   768
#define DEPTH 6
#define HEADS 12
#define DHEAD 64
#define MLPD  3072
#define D3    2304        // 3*DIM
#define PIX   768         // 16*16*3
#define BATCH 64
#define NMASK 98          // int(0.5*196)
#define MROWS (BATCH*NTOK)    // 12544 = 49*256
#define PROWS (BATCH*NMASK)   // 6272
#define MPAD  6400            // 25*256, padded rows for pixel-head GEMM

typedef __bf16 bf16_t;
typedef bf16_t v16bf __attribute__((ext_vector_type(16)));
typedef float  v8f   __attribute__((ext_vector_type(8)));

// ---------------- device helpers ----------------
__device__ __forceinline__ float wave_sum(float v) {
#pragma unroll
  for (int o = 16; o > 0; o >>= 1) v += __shfl_xor(v, o, 32);
  return v;
}
__device__ __forceinline__ float wave_max(float v) {
#pragma unroll
  for (int o = 16; o > 0; o >>= 1) v = fmaxf(v, __shfl_xor(v, o, 32));
  return v;
}
__device__ __forceinline__ float gelu_tanh(float x) {
  float x3 = x * x * x;
  return 0.5f * x * (1.0f + tanhf(0.7978845608028654f * (x + 0.044715f * x3)));
}

// ---------------- WMMA fragment maps (ISA 7.12.2) ----------------
// A-frag (16x32 bf16): lane = half*16 + (m&15); element e -> r=e>>1,
//   k' = (r&4?16:0) + 2*(r&3) + 8*half + (e&1)
// Inverse (given k'=k&31): half = (k'>>3)&1 ; e = ((k'&16)>>1)|(k'&6)|(k'&1)
// Packed A: ((mt*KT + kt)*512 + lane*16 + e)
// B-frag (32x16 bf16): lane = half*16 + n ; e -> k' = (e&8?16:0) + 8*half + (e&7)
// Packed B: ((kt*NT + nt)*512 + lane*16 + e)

__device__ __forceinline__ size_t packA_index(int m, int k, int K) {
  int kt = k >> 5, kp = k & 31;
  int half = (kp >> 3) & 1;
  int e = ((kp & 16) >> 1) | (kp & 6) | (kp & 1);
  int lane = half * 16 + (m & 15);
  return (((size_t)(m >> 4) * (K >> 5) + kt) * 32 + lane) * 16 + e;
}

// ---------------- weight packing (B-frag layout) ----------------
__global__ __launch_bounds__(256) void simmim_pack_b_f32(
    const float* __restrict__ B, bf16_t* __restrict__ out, int K, int N) {
  size_t idx = (size_t)blockIdx.x * 256 + threadIdx.x;
  size_t total = (size_t)K * N;
  if (idx >= total) return;
  int e = idx & 15;
  int lane = (idx >> 4) & 31;
  size_t t = idx >> 9;
  int ntiles = N >> 4;
  int nt = (int)(t % ntiles);
  int kt = (int)(t / ntiles);
  int k = kt * 32 + ((e & 8) << 1) + 8 * (lane >> 4) + (e & 7);
  int n = nt * 16 + (lane & 15);
  out[idx] = (bf16_t)B[(size_t)k * N + n];
}

// patchify + pack directly from the image into A-frag layout
__global__ __launch_bounds__(256) void simmim_pack_patches(
    const float* __restrict__ img, bf16_t* __restrict__ out) {
  size_t idx = (size_t)blockIdx.x * 256 + threadIdx.x;
  if (idx >= (size_t)MROWS * PIX) return;
  int e = idx & 15;
  int lane = (idx >> 4) & 31;
  size_t t = idx >> 9;
  const int KT = PIX / 32;  // 24
  int kt = (int)(t % KT);
  int mt = (int)(t / KT);
  int half = lane >> 4;
  int r = e >> 1;
  int k = kt * 32 + ((r & 4) ? 16 : 0) + ((r & 3) << 1) + 8 * half + (e & 1);
  int m = mt * 16 + (lane & 15);
  int b = m / NTOK, i = m % NTOK;
  int gy = i / PGRID, gx = i % PGRID;
  int c = k % 3;
  int pxy = k / 3;
  int px = pxy & 15, py = pxy >> 4;
  out[idx] = (bf16_t)img[(((size_t)(b * IMGSZ + gy * PATCH + py)) * IMGSZ +
                          (gx * PATCH + px)) * 3 + c];
}

// ---------------- WMMA GEMM ----------------
// grid: (N/64, M/256), block 256 (8 waves). Wave -> 32x64 strip (2 A-frags x 4 B-frags).
// Software-pipelined: fragments for kt+1 are loaded before the wmma block on kt.
// Output: packed-A bf16 (Cp, for chained GEMM) or f32 row-major (Cf), with
// fused bias / tanh-GELU / residual.
__global__ __launch_bounds__(256) void simmim_gemm_bf16_wmma(
    const bf16_t* __restrict__ pA, const bf16_t* __restrict__ pB,
    const float* __restrict__ bias, const float* __restrict__ resid,
    float* __restrict__ Cf, bf16_t* __restrict__ Cp,
    int M, int N, int K, int act) {
  const int lane = threadIdx.x & 31;
  const int wv = threadIdx.x >> 5;
  const int ktiles = K >> 5;
  const int ntiles = N >> 4;
  const int mt0 = (blockIdx.y * 8 + wv) * 2;
  const int col0 = blockIdx.x * 64;
  const int tn0 = blockIdx.x * 4;

  v8f acc[2][4] = {};
  const bf16_t* aP0 = pA + ((size_t)mt0 * ktiles) * 512 + (size_t)lane * 16;
  const bf16_t* aP1 = aP0 + (size_t)ktiles * 512;
  const size_t bstride = (size_t)ntiles * 512;
  const bf16_t* bp = pB + ((size_t)tn0 * 32 + lane) * 16;

  v16bf a0 = *(const v16bf*)(aP0);
  v16bf a1 = *(const v16bf*)(aP1);
  v16bf b0 = *(const v16bf*)(bp);
  v16bf b1 = *(const v16bf*)(bp + 512);
  v16bf b2 = *(const v16bf*)(bp + 1024);
  v16bf b3 = *(const v16bf*)(bp + 1536);

  for (int kt = 0; kt < ktiles - 1; ++kt) {
    const bf16_t* ap0n = aP0 + (size_t)(kt + 1) * 512;
    const bf16_t* ap1n = aP1 + (size_t)(kt + 1) * 512;
    const bf16_t* bpn = bp + (size_t)(kt + 1) * bstride;
    v16bf na0 = *(const v16bf*)(ap0n);
    v16bf na1 = *(const v16bf*)(ap1n);
    v16bf nb0 = *(const v16bf*)(bpn);
    v16bf nb1 = *(const v16bf*)(bpn + 512);
    v16bf nb2 = *(const v16bf*)(bpn + 1024);
    v16bf nb3 = *(const v16bf*)(bpn + 1536);
    if (kt + 5 < ktiles)
      __builtin_prefetch(bp + (size_t)(kt + 5) * bstride, 0, 1);

    acc[0][0] = __builtin_amdgcn_wmma_f32_16x16x32_bf16(false, a0, false, b0, (short)0, acc[0][0], false, false);
    acc[0][1] = __builtin_amdgcn_wmma_f32_16x16x32_bf16(false, a0, false, b1, (short)0, acc[0][1], false, false);
    acc[0][2] = __builtin_amdgcn_wmma_f32_16x16x32_bf16(false, a0, false, b2, (short)0, acc[0][2], false, false);
    acc[0][3] = __builtin_amdgcn_wmma_f32_16x16x32_bf16(false, a0, false, b3, (short)0, acc[0][3], false, false);
    acc[1][0] = __builtin_amdgcn_wmma_f32_16x16x32_bf16(false, a1, false, b0, (short)0, acc[1][0], false, false);
    acc[1][1] = __builtin_amdgcn_wmma_f32_16x16x32_bf16(false, a1, false, b1, (short)0, acc[1][1], false, false);
    acc[1][2] = __builtin_amdgcn_wmma_f32_16x16x32_bf16(false, a1, false, b2, (short)0, acc[1][2], false, false);
    acc[1][3] = __builtin_amdgcn_wmma_f32_16x16x32_bf16(false, a1, false, b3, (short)0, acc[1][3], false, false);

    a0 = na0; a1 = na1; b0 = nb0; b1 = nb1; b2 = nb2; b3 = nb3;
  }
  // final K-step
  acc[0][0] = __builtin_amdgcn_wmma_f32_16x16x32_bf16(false, a0, false, b0, (short)0, acc[0][0], false, false);
  acc[0][1] = __builtin_amdgcn_wmma_f32_16x16x32_bf16(false, a0, false, b1, (short)0, acc[0][1], false, false);
  acc[0][2] = __builtin_amdgcn_wmma_f32_16x16x32_bf16(false, a0, false, b2, (short)0, acc[0][2], false, false);
  acc[0][3] = __builtin_amdgcn_wmma_f32_16x16x32_bf16(false, a0, false, b3, (short)0, acc[0][3], false, false);
  acc[1][0] = __builtin_amdgcn_wmma_f32_16x16x32_bf16(false, a1, false, b0, (short)0, acc[1][0], false, false);
  acc[1][1] = __builtin_amdgcn_wmma_f32_16x16x32_bf16(false, a1, false, b1, (short)0, acc[1][1], false, false);
  acc[1][2] = __builtin_amdgcn_wmma_f32_16x16x32_bf16(false, a1, false, b2, (short)0, acc[1][2], false, false);
  acc[1][3] = __builtin_amdgcn_wmma_f32_16x16x32_bf16(false, a1, false, b3, (short)0, acc[1][3], false, false);

  const int half = lane >> 4;
  const int nl = lane & 15;
#pragma unroll
  for (int mi = 0; mi < 2; ++mi) {
#pragma unroll
    for (int j = 0; j < 4; ++j) {
#pragma unroll
      for (int r = 0; r < 8; ++r) {
        int row = (mt0 + mi) * 16 + r + 8 * half;
        int col = col0 + j * 16 + nl;
        float v = acc[mi][j][r];
        if (bias) v += bias[col];
        if (act == 1) v = gelu_tanh(v);
        if (Cp) {
          Cp[packA_index(row, col, N)] = (bf16_t)v;
        } else {
          if (resid) v += resid[(size_t)row * N + col];
          Cf[(size_t)row * N + col] = v;
        }
      }
    }
  }
}

// ---------------- masking ----------------
__global__ void simmim_mask_flags(const float* __restrict__ noise, int* __restrict__ flags) {
  int idx = blockIdx.x * 256 + threadIdx.x;
  if (idx >= BATCH * NTOK) return;
  int b = idx / NTOK, i = idx % NTOK;
  const float* nr = noise + (size_t)b * NTOK;
  float ni = nr[i];
  int rank = 0;
  for (int j = 0; j < NTOK; ++j) {
    float nj = nr[j];
    rank += (nj > ni) || (nj == ni && j < i);
  }
  flags[idx] = (rank < NMASK) ? 1 : 0;
}

__global__ void simmim_mask_slots(const int* __restrict__ flags, int* __restrict__ midx) {
  int idx = blockIdx.x * 256 + threadIdx.x;
  if (idx >= BATCH * NTOK) return;
  int b = idx / NTOK, i = idx % NTOK;
  if (!flags[idx]) return;
  const int* fr = flags + (size_t)b * NTOK;
  int slot = 0;
  for (int j = 0; j < i; ++j) slot += fr[j];
  midx[b * NMASK + slot] = i;
}

// X = masked ? mask_token+pe : X+pe
__global__ void simmim_tokens_finalize(float* __restrict__ X,
                                       const float* __restrict__ pos_emb,
                                       const float* __restrict__ mask_token,
                                       const int* __restrict__ flags) {
  size_t idx = (size_t)blockIdx.x * 256 + threadIdx.x;
  if (idx >= (size_t)MROWS * DIM) return;
  int d = (int)(idx % DIM);
  size_t row = idx / DIM;
  int i = (int)(row % NTOK);
  float pe = pos_emb[(size_t)(i + 1) * DIM + d];
  X[idx] = flags[row] ? (mask_token[d] + pe) : (X[idx] + pe);
}

// ---------------- layernorm, writing packed-A bf16 directly ----------------
__global__ __launch_bounds__(256) void simmim_layernorm_pack(
    const float* __restrict__ X, const float* __restrict__ g,
    const float* __restrict__ bb, bf16_t* __restrict__ out) {
  int row = blockIdx.x;
  int t = threadIdx.x;
  __shared__ float redA[8], redB[8];
  __shared__ float s_mean, s_rstd;
  const float* xr = X + (size_t)row * DIM;
  float s = 0.f, sq = 0.f;
  for (int d = t; d < DIM; d += 256) {
    float v = xr[d];
    s += v;
    sq += v * v;
  }
  s = wave_sum(s);
  sq = wave_sum(sq);
  if ((t & 31) == 0) { redA[t >> 5] = s; redB[t >> 5] = sq; }
  __syncthreads();
  if (t == 0) {
    float a = 0.f, b2 = 0.f;
    for (int i = 0; i < 8; ++i) { a += redA[i]; b2 += redB[i]; }
    float mean = a / DIM;
    float var = b2 / DIM - mean * mean;
    s_mean = mean;
    s_rstd = rsqrtf(var + 1e-5f);
  }
  __syncthreads();
  float mean = s_mean, rstd = s_rstd;
  for (int d = t; d < DIM; d += 256) {
    float v = (xr[d] - mean) * rstd * g[d] + bb[d];
    out[packA_index(row, d, DIM)] = (bf16_t)v;
  }
}

// ---------------- attention (one block per (b,h)); output packed-A ----------------
__global__ __launch_bounds__(256) void simmim_attention(
    const float* __restrict__ qkv, bf16_t* __restrict__ outp) {
  int bh = blockIdx.x;
  int b = bh / HEADS, h = bh % HEADS;
  int t = threadIdx.x;
  __shared__ float qrow[DHEAD];
  __shared__ float sc[NTOK];
  __shared__ float red[8];
  __shared__ float sh_v;
  const float scale = 0.125f;  // 1/sqrt(64)
  size_t base = (size_t)b * NTOK * D3;
  for (int n = 0; n < NTOK; ++n) {
    if (t < DHEAD) qrow[t] = qkv[base + (size_t)n * D3 + h * DHEAD + t];
    __syncthreads();
    float s = -3.4e38f;
    if (t < NTOK) {
      const float* kr = qkv + base + (size_t)t * D3 + DIM + h * DHEAD;
      float acc = 0.f;
#pragma unroll 8
      for (int d = 0; d < DHEAD; ++d) acc += qrow[d] * kr[d];
      s = acc * scale;
    }
    float m = wave_max(s);
    if ((t & 31) == 0) red[t >> 5] = m;
    __syncthreads();
    if (t == 0) {
      float mm = red[0];
      for (int i = 1; i < 8; ++i) mm = fmaxf(mm, red[i]);
      sh_v = mm;
    }
    __syncthreads();
    float mx = sh_v;
    float p = (t < NTOK) ? __expf(s - mx) : 0.f;
    float su = wave_sum(p);
    if ((t & 31) == 0) red[t >> 5] = su;
    __syncthreads();
    if (t == 0) {
      float ss = red[0];
      for (int i = 1; i < 8; ++i) ss += red[i];
      sh_v = ss;
    }
    __syncthreads();
    float inv = 1.0f / sh_v;
    if (t < NTOK) sc[t] = p * inv;
    __syncthreads();
    if (t < DHEAD) {
      float acc = 0.f;
      const float* vb = qkv + base + 2 * DIM + h * DHEAD + t;
      for (int j = 0; j < NTOK; ++j) acc += sc[j] * vb[(size_t)j * D3];
      int mrow = b * NTOK + n;
      int k = h * DHEAD + t;
      outp[packA_index(mrow, k, DIM)] = (bf16_t)acc;
    }
    __syncthreads();
  }
}

// ---------------- gather masked rows into packed-A (padded to MPAD rows) ----------------
__global__ void simmim_gather_pack(const float* __restrict__ X,
                                   const int* __restrict__ midx,
                                   bf16_t* __restrict__ out) {
  size_t idx = (size_t)blockIdx.x * 256 + threadIdx.x;
  if (idx >= (size_t)MPAD * DIM) return;
  int e = idx & 15;
  int lane = (idx >> 4) & 31;
  size_t t = idx >> 9;
  const int KT = DIM >> 5;  // 24
  int kt = (int)(t % KT);
  int mt = (int)(t / KT);
  int half = lane >> 4;
  int r = e >> 1;
  int k = kt * 32 + ((r & 4) ? 16 : 0) + ((r & 3) << 1) + 8 * half + (e & 1);
  int m = mt * 16 + (lane & 15);
  float v = 0.f;
  if (m < PROWS) {
    int b = m / NMASK;
    int i = midx[m];
    v = X[((size_t)b * NTOK + i) * DIM + k];
  }
  out[idx] = (bf16_t)v;
}

// ---------------- loss ----------------
__global__ __launch_bounds__(256) void simmim_loss_partial(
    const float* __restrict__ pred, const float* __restrict__ img,
    const int* __restrict__ midx, float* __restrict__ partials) {
  size_t idx = (size_t)blockIdx.x * 256 + threadIdx.x;  // total = PROWS*PIX exactly
  int p = (int)(idx % PIX);
  size_t r = idx / PIX;
  int b = (int)(r / NMASK);
  int i = midx[r];
  int gy = i / PGRID, gx = i % PGRID;
  int c = p % 3;
  int pxy = p / 3;
  int px = pxy & 15, py = pxy >> 4;
  float tv = img[(((size_t)(b * IMGSZ + gy * PATCH + py)) * IMGSZ +
                  (gx * PATCH + px)) * 3 + c];
  float d = fabsf(pred[idx] - tv);
  __shared__ float red[8];
  float s = wave_sum(d);
  if ((threadIdx.x & 31) == 0) red[threadIdx.x >> 5] = s;
  __syncthreads();
  if (threadIdx.x == 0) {
    float a = 0.f;
    for (int k = 0; k < 8; ++k) a += red[k];
    partials[blockIdx.x] = a;
  }
}

__global__ __launch_bounds__(256) void simmim_loss_final(
    const float* __restrict__ partials, float* __restrict__ out, int nb) {
  float s = 0.f;
  for (int i = threadIdx.x; i < nb; i += 256) s += partials[i];
  s = wave_sum(s);
  __shared__ float red[8];
  if ((threadIdx.x & 31) == 0) red[threadIdx.x >> 5] = s;
  __syncthreads();
  if (threadIdx.x == 0) {
    float a = 0.f;
    for (int k = 0; k < 8; ++k) a += red[k];
    // mean over (64,98,768) then / NUM_MASKED
    out[0] = a * (1.0f / 4816896.0f) / 98.0f;
  }
}

// ---------------- host launcher ----------------
static inline size_t alignup(size_t x) { return (x + 255) & ~(size_t)255; }

extern "C" void kernel_launch(void* const* d_in, const int* in_sizes, int n_in,
                              void* d_out, int out_size, void* d_ws, size_t ws_size,
                              hipStream_t stream) {
  (void)in_sizes; (void)n_in; (void)out_size; (void)ws_size;
  const float* img        = (const float*)d_in[0];
  const float* mask_noise = (const float*)d_in[1];
  const float* patch_w    = (const float*)d_in[2];
  const float* patch_b    = (const float*)d_in[3];
  const float* pos_emb    = (const float*)d_in[4];
  const float* mask_token = (const float*)d_in[5];
  const float* ln1_s      = (const float*)d_in[6];
  const float* ln1_b      = (const float*)d_in[7];
  const float* wqkv       = (const float*)d_in[8];
  const float* bqkv       = (const float*)d_in[9];
  const float* wo         = (const float*)d_in[10];
  const float* bo         = (const float*)d_in[11];
  const float* ln2_s      = (const float*)d_in[12];
  const float* ln2_b      = (const float*)d_in[13];
  const float* w1         = (const float*)d_in[14];
  const float* b1         = (const float*)d_in[15];
  const float* w2         = (const float*)d_in[16];
  const float* b2         = (const float*)d_in[17];
  const float* pix_w      = (const float*)d_in[18];
  const float* pix_b      = (const float*)d_in[19];
  float* out = (float*)d_out;

  // workspace layout
  char* ws = (char*)d_ws;
  size_t off = 0;
  float*  X    = (float*)(ws + off);  off = alignup(off + (size_t)MROWS * DIM * 4);
  float*  QKV  = (float*)(ws + off);  off = alignup(off + (size_t)MROWS * D3 * 4);
  bf16_t* pA   = (bf16_t*)(ws + off); off = alignup(off + (size_t)MROWS * DIM * 2);
  bf16_t* pG   = (bf16_t*)(ws + off); off = alignup(off + (size_t)MROWS * MLPD * 2);
  bf16_t* pW   = (bf16_t*)(ws + off); off = alignup(off + (size_t)DIM * MLPD * 2);
  int*    flags= (int*)(ws + off);    off = alignup(off + (size_t)BATCH * NTOK * 4);
  int*    midx = (int*)(ws + off);    off = alignup(off + (size_t)BATCH * NMASK * 4);
  float* partials = (float*)(ws + off);
  float* PRED = QKV;  // pixel-head output reuses the (idle) QKV buffer

  const int T = 256;
  auto blocks = [](size_t n) { return (unsigned)((n + 255) / 256); };

  // masking
  simmim_mask_flags<<<blocks((size_t)BATCH * NTOK), T, 0, stream>>>(mask_noise, flags);
  simmim_mask_slots<<<blocks((size_t)BATCH * NTOK), T, 0, stream>>>(flags, midx);

  // patch embed GEMM + pos_emb/mask substitution
  simmim_pack_patches<<<blocks((size_t)MROWS * PIX), T, 0, stream>>>(img, pA);
  simmim_pack_b_f32<<<blocks((size_t)PIX * DIM), T, 0, stream>>>(patch_w, pW, PIX, DIM);
  simmim_gemm_bf16_wmma<<<dim3(DIM / 64, MROWS / 256), T, 0, stream>>>(
      pA, pW, patch_b, nullptr, X, nullptr, MROWS, DIM, PIX, 0);
  simmim_tokens_finalize<<<blocks((size_t)MROWS * DIM), T, 0, stream>>>(
      X, pos_emb, mask_token, flags);

  // transformer layers
  for (int l = 0; l < DEPTH; ++l) {
    // LN1 (packed) -> QKV GEMM
    simmim_layernorm_pack<<<MROWS, T, 0, stream>>>(X, ln1_s + (size_t)l * DIM,
                                                   ln1_b + (size_t)l * DIM, pA);
    simmim_pack_b_f32<<<blocks((size_t)DIM * D3), T, 0, stream>>>(
        wqkv + (size_t)l * DIM * D3, pW, DIM, D3);
    simmim_gemm_bf16_wmma<<<dim3(D3 / 64, MROWS / 256), T, 0, stream>>>(
        pA, pW, bqkv + (size_t)l * D3, nullptr, QKV, nullptr, MROWS, D3, DIM, 0);

    // attention -> packed-A
    simmim_attention<<<BATCH * HEADS, T, 0, stream>>>(QKV, pA);

    // x = x + attn @ Wo + bo
    simmim_pack_b_f32<<<blocks((size_t)DIM * DIM), T, 0, stream>>>(
        wo + (size_t)l * DIM * DIM, pW, DIM, DIM);
    simmim_gemm_bf16_wmma<<<dim3(DIM / 64, MROWS / 256), T, 0, stream>>>(
        pA, pW, bo + (size_t)l * DIM, X, X, nullptr, MROWS, DIM, DIM, 0);

    // LN2 (packed) -> fc1 GEMM (gelu, packed out) -> fc2 GEMM (+resid)
    simmim_layernorm_pack<<<MROWS, T, 0, stream>>>(X, ln2_s + (size_t)l * DIM,
                                                   ln2_b + (size_t)l * DIM, pA);
    simmim_pack_b_f32<<<blocks((size_t)DIM * MLPD), T, 0, stream>>>(
        w1 + (size_t)l * DIM * MLPD, pW, DIM, MLPD);
    simmim_gemm_bf16_wmma<<<dim3(MLPD / 64, MROWS / 256), T, 0, stream>>>(
        pA, pW, b1 + (size_t)l * MLPD, nullptr, nullptr, pG, MROWS, MLPD, DIM, 1);
    simmim_pack_b_f32<<<blocks((size_t)MLPD * DIM), T, 0, stream>>>(
        w2 + (size_t)l * MLPD * DIM, pW, MLPD, DIM);
    simmim_gemm_bf16_wmma<<<dim3(DIM / 64, MROWS / 256), T, 0, stream>>>(
        pG, pW, b2 + (size_t)l * DIM, X, X, nullptr, MROWS, DIM, MLPD, 0);
  }

  // gather masked tokens (packed, padded) -> pixel head GEMM
  simmim_gather_pack<<<blocks((size_t)MPAD * DIM), T, 0, stream>>>(X, midx, pA);
  simmim_pack_b_f32<<<blocks((size_t)DIM * PIX), T, 0, stream>>>(pix_w, pW, DIM, PIX);
  simmim_gemm_bf16_wmma<<<dim3(PIX / 64, MPAD / 256), T, 0, stream>>>(
      pA, pW, pix_b, nullptr, PRED, nullptr, MPAD, PIX, DIM, 0);

  // L1 loss, deterministic two-stage reduction
  const int nb = (int)(((size_t)PROWS * PIX) / 256);  // 18816
  simmim_loss_partial<<<nb, T, 0, stream>>>(PRED, img, midx, partials);
  simmim_loss_final<<<1, T, 0, stream>>>(partials, out, nb);
}
